// TransformerBlock_76605036691609
// MI455X (gfx1250) — compile-verified, matrix operand order checked by reading, not run on hardware
//
#include <hip/hip_runtime.h>

// ---------------------------------------------------------------------------
// Transformer block for MI455X (gfx1250, wave32, WMMA).
// All matmuls run on V_WMMA_F32_16X16X32_BF16 (fp32 accumulate).
// Tile staging uses GLOBAL_LOAD_ASYNC_TO_LDS (ASYNCcnt) when available.
// ---------------------------------------------------------------------------

typedef unsigned short bfraw; // raw bf16 bits
typedef __attribute__((ext_vector_type(16))) __bf16 bf16x16;
typedef __attribute__((ext_vector_type(8)))  float  f32x8;
typedef __attribute__((__vector_size__(16))) int i32x4v;  // builtin pointee type

union Frag { bf16x16 v; uint4 q[2]; };

#define SEQ    2048
#define TOKENS 4096   // B*L
#define DMODEL 1024
#define DFF    4096
#define NQKV   3072

#define AS_GLOBAL __attribute__((address_space(1)))
#define AS_LDS    __attribute__((address_space(3)))

#if defined(__HIP_DEVICE_COMPILE__) && \
    __has_builtin(__builtin_amdgcn_global_load_async_to_lds_b128) && \
    __has_builtin(__builtin_amdgcn_s_wait_asynccnt)
#define USE_ASYNC 1
#else
#define USE_ASYNC 0
#endif

__device__ inline bfraw f2bf(float f) {
  unsigned int u = __float_as_uint(f);
  u += 0x7FFFu + ((u >> 16) & 1u);          // round-to-nearest-even
  return (bfraw)(u >> 16);
}
__device__ inline float bf2f(bfraw h) {
  return __uint_as_float(((unsigned int)h) << 16);
}
__device__ inline f32x8 zero8() {
  f32x8 z;
#pragma unroll
  for (int i = 0; i < 8; ++i) z[i] = 0.0f;
  return z;
}

// 16-byte global -> LDS copy: async DMA path on CDNA5, manual fallback.
__device__ inline void cp16(const bfraw* gsrc, bfraw* ldst) {
#if USE_ASYNC
  __builtin_amdgcn_global_load_async_to_lds_b128(
      (AS_GLOBAL i32x4v*)gsrc, (AS_LDS i32x4v*)ldst, 0, 0);
#else
  *(uint4*)ldst = *(const uint4*)gsrc;
#endif
}
__device__ inline void cp_wait() {
#if USE_ASYNC
  __builtin_amdgcn_s_wait_asynccnt(0);
#endif
}

// ---------------------------------------------------------------------------
// Weight cast + transpose: fp32 [K][N] row-major -> bf16 [N][K] row-major.
// ---------------------------------------------------------------------------
__global__ __launch_bounds__(256) void transpose_cast_kernel(
    const float* __restrict__ in, bfraw* __restrict__ out, int K, int N) {
  __shared__ float tile[32][33];
  int tx = threadIdx.x & 31, ty = threadIdx.x >> 5;
  int n0 = blockIdx.x * 32, k0 = blockIdx.y * 32;
#pragma unroll
  for (int i = 0; i < 32; i += 8)
    tile[ty + i][tx] = in[(size_t)(k0 + ty + i) * N + (n0 + tx)];
  __syncthreads();
#pragma unroll
  for (int i = 0; i < 32; i += 8)
    out[(size_t)(n0 + ty + i) * K + (k0 + tx)] = f2bf(tile[tx][ty + i]);
}

// ---------------------------------------------------------------------------
// RMSNorm: fp32 row -> bf16 row.  One 256-thread block per row (D=1024).
// ---------------------------------------------------------------------------
__global__ __launch_bounds__(256) void rmsnorm_kernel(
    const float* __restrict__ x, const float* __restrict__ g,
    bfraw* __restrict__ out) {
  int row = blockIdx.x;
  int tid = threadIdx.x;
  const float* xr = x + (size_t)row * DMODEL;
  float4 xv = *(const float4*)(xr + tid * 4);
  float ss = xv.x * xv.x + xv.y * xv.y + xv.z * xv.z + xv.w * xv.w;
#pragma unroll
  for (int off = 16; off > 0; off >>= 1) ss += __shfl_xor(ss, off, 32);
  __shared__ float red[8];
  if ((tid & 31) == 0) red[tid >> 5] = ss;
  __syncthreads();
  float tot = 0.0f;
#pragma unroll
  for (int i = 0; i < 8; ++i) tot += red[i];
  float inv = rsqrtf(tot * (1.0f / DMODEL) + 1e-6f);
  float4 gv = *(const float4*)(g + tid * 4);
  bfraw* o = out + (size_t)row * DMODEL + tid * 4;
  o[0] = f2bf(xv.x * inv * gv.x);
  o[1] = f2bf(xv.y * inv * gv.y);
  o[2] = f2bf(xv.z * inv * gv.z);
  o[3] = f2bf(xv.w * inv * gv.w);
}

// ---------------------------------------------------------------------------
// WMMA GEMM: C[M,N] = A[M,K](bf16, row-major) x Bw[N,K](bf16, N-major = B^T).
// 128x128 tile per 256-thread workgroup; 8 waves in a 2x4 grid, each wave
// owns 64x32 (4x2 WMMA tiles).  K-step 32, double-buffered LDS staging via
// async global->LDS copies overlapped with WMMA compute.
// ---------------------------------------------------------------------------
enum { MODE_BF16 = 0, MODE_RES_F32 = 1, MODE_F32 = 2, MODE_SILU_BF16 = 3 };

template <int MODE>
__global__ __launch_bounds__(256) void gemm_bf16_kernel(
    const bfraw* __restrict__ A, const bfraw* __restrict__ Bw,
    const float* __restrict__ res, const float* __restrict__ gate,
    float* __restrict__ outf, bfraw* __restrict__ outb,
    int M, int N, int K) {
  __shared__ __align__(16) bfraw As[2][128 * 40];
  __shared__ __align__(16) bfraw Bs[2][128 * 40];
  int tid = threadIdx.x;
  int lane = tid & 31;
  int wid = tid >> 5;
  int wm = wid & 1;        // 0..1 -> 64-row stripe
  int wn = wid >> 1;       // 0..3 -> 32-col stripe
  int lr = lane & 15;
  int kb = (lane >> 4) << 3;      // 0 or 8 (per-half-wave K base)
  int rowBase = blockIdx.y * 128;
  int colBase = blockIdx.x * 128;

  // stage one 128x32 A tile + 128x32 B tile into LDS buffer `buf`
  auto stage = [&](int buf, int k0) {
    for (int i = tid; i < 512; i += 256) {
      int r = i >> 2, seg = i & 3;
      cp16(A + (size_t)(rowBase + r) * K + k0 + seg * 8, &As[buf][r * 40 + seg * 8]);
      cp16(Bw + (size_t)(colBase + r) * K + k0 + seg * 8, &Bs[buf][r * 40 + seg * 8]);
    }
  };

  f32x8 acc[4][2];
#pragma unroll
  for (int mt = 0; mt < 4; ++mt)
#pragma unroll
    for (int nt = 0; nt < 2; ++nt) acc[mt][nt] = zero8();

  int nk = K >> 5;
  stage(0, 0);
  cp_wait();
  __syncthreads();

  for (int kt = 0; kt < nk; ++kt) {
    int cur = kt & 1;
    if (kt + 1 < nk) stage(cur ^ 1, (kt + 1) << 5);   // overlap with compute
    if (kt + 2 < nk) {                                // warm L2 two tiles out
      int kp = (kt + 2) << 5;
      __builtin_prefetch(A + (size_t)(rowBase + (tid >> 1)) * K + kp + (tid & 1) * 16, 0, 1);
      __builtin_prefetch(Bw + (size_t)(colBase + (tid >> 1)) * K + kp + (tid & 1) * 16, 0, 1);
    }

    Frag bf[2];
#pragma unroll
    for (int nt = 0; nt < 2; ++nt) {
      int base = (wn * 32 + nt * 16 + lr) * 40 + kb;
      bf[nt].q[0] = *(const uint4*)&Bs[cur][base];
      bf[nt].q[1] = *(const uint4*)&Bs[cur][base + 16];
    }
#pragma unroll
    for (int mt = 0; mt < 4; ++mt) {
      Frag af;
      int base = (wm * 64 + mt * 16 + lr) * 40 + kb;
      af.q[0] = *(const uint4*)&As[cur][base];
      af.q[1] = *(const uint4*)&As[cur][base + 16];
#pragma unroll
      for (int nt = 0; nt < 2; ++nt)
        acc[mt][nt] = __builtin_amdgcn_wmma_f32_16x16x32_bf16(
            false, af.v, false, bf[nt].v, (short)0, acc[mt][nt], false, false);
    }
    cp_wait();
    __syncthreads();
  }

  // ---- epilogue: C layout lane->(m=(lane/16)*8+r, n=lane%16) ----
#pragma unroll
  for (int mt = 0; mt < 4; ++mt) {
#pragma unroll
    for (int nt = 0; nt < 2; ++nt) {
#pragma unroll
      for (int r = 0; r < 8; ++r) {
        int row = rowBase + wm * 64 + mt * 16 + ((lane >> 4) << 3) + r;
        int col = colBase + wn * 32 + nt * 16 + lr;
        size_t idx = (size_t)row * N + col;
        float v = acc[mt][nt][r];
        if constexpr (MODE == MODE_BF16) {
          outb[idx] = f2bf(v);
        } else if constexpr (MODE == MODE_RES_F32) {
          outf[idx] = v + res[idx];
        } else if constexpr (MODE == MODE_F32) {
          outf[idx] = v;
        } else {  // MODE_SILU_BF16: acc is "up", gate buffer holds fp32 gate
          float gv = gate[idx];
          float s = gv / (1.0f + __expf(-gv));
          outb[idx] = f2bf(s * v);
        }
      }
    }
  }
}

// ---------------------------------------------------------------------------
// Flash attention: one block = 128 q-rows of one (b,h); 8 waves x 16 q-rows.
// S = (Q/sqrt(dk)) K^T via WMMA, online softmax (shfl row reductions),
// P restaged through per-wave LDS, O += P V via WMMA.
// K tile staged with async global->LDS; V needs a transpose scatter.
// ---------------------------------------------------------------------------
__global__ __launch_bounds__(256) void attn_kernel(
    const bfraw* __restrict__ qkv, bfraw* __restrict__ out) {
  __shared__ __align__(16) bfraw Qs[128 * 72];   // [qrow][feat], pad 72
  __shared__ __align__(16) bfraw Ks[32 * 72];    // [key][feat]
  __shared__ __align__(16) bfraw Vt[64 * 40];    // [feat][key] (transposed)
  __shared__ __align__(16) bfraw Ps[8][16 * 40]; // per-wave P staging

  int tid = threadIdx.x;
  int lane = tid & 31;
  int wid = tid >> 5;
  int lr = lane & 15;
  int kb = (lane >> 4) << 3;
  int mrow = (lane >> 4) << 3;  // + r -> local q row in 16x16 C tile
  int bh = blockIdx.y;
  int b = bh >> 4, h = bh & 15;
  int qbase = blockIdx.x * 128;
  const float scale = 0.125f;  // 1/sqrt(64)

  // ---- load Q tile (scaled), vectorized 8 bf16 per thread-chunk ----
  for (int i = tid; i < 1024; i += 256) {
    int r = i >> 3, c = (i & 7) * 8;
    size_t tok = (size_t)b * SEQ + qbase + r;
    uint4 qq = *(const uint4*)(qkv + tok * NQKV + h * 64 + c);
    __align__(16) bfraw t8[8];
    *(uint4*)t8 = qq;
    __align__(16) bfraw o8[8];
#pragma unroll
    for (int j = 0; j < 8; ++j) o8[j] = f2bf(bf2f(t8[j]) * scale);
    *(uint4*)&Qs[r * 72 + c] = *(uint4*)o8;
  }

  f32x8 o[4];
#pragma unroll
  for (int ft = 0; ft < 4; ++ft) o[ft] = zero8();
  float row_max[8], row_sum[8];
#pragma unroll
  for (int r = 0; r < 8; ++r) { row_max[r] = -1e30f; row_sum[r] = 0.0f; }

  int qr0 = qbase + wid * 16;
  int nkt = (qbase + 128) >> 5;  // causal: key tiles 0 .. qbase/32 + 3

  for (int kt = 0; kt < nkt; ++kt) {
    __syncthreads();  // previous iteration's Ks/Vt/Ps reads complete
    {  // stage K (async, row-major) and V (transposed) tiles: 32 keys x 64 f
      int r = tid >> 3, c = (tid & 7) * 8;
      size_t tok = (size_t)b * SEQ + kt * 32 + r;
      cp16(qkv + tok * NQKV + 1024 + h * 64 + c, &Ks[r * 72 + c]);
      uint4 vq = *(const uint4*)(qkv + tok * NQKV + 2048 + h * 64 + c);
      __align__(16) bfraw t8[8];
      *(uint4*)t8 = vq;
#pragma unroll
      for (int j = 0; j < 8; ++j) Vt[(c + j) * 40 + r] = t8[j];
    }
    cp_wait();
    __syncthreads();

    // ---- S[16 x 32] = Q[16 x 64] * K^T : two 16-col tiles, K-dim 2x32 ----
    f32x8 s0 = zero8(), s1 = zero8();
#pragma unroll
    for (int kk = 0; kk < 64; kk += 32) {
      Frag qf;
      int ab = (wid * 16 + lr) * 72 + kk + kb;
      qf.q[0] = *(const uint4*)&Qs[ab];
      qf.q[1] = *(const uint4*)&Qs[ab + 16];
      Frag kf0, kf1;
      int b0 = lr * 72 + kk + kb;
      int b1 = (16 + lr) * 72 + kk + kb;
      kf0.q[0] = *(const uint4*)&Ks[b0];
      kf0.q[1] = *(const uint4*)&Ks[b0 + 16];
      kf1.q[0] = *(const uint4*)&Ks[b1];
      kf1.q[1] = *(const uint4*)&Ks[b1 + 16];
      s0 = __builtin_amdgcn_wmma_f32_16x16x32_bf16(false, qf.v, false, kf0.v,
                                                   (short)0, s0, false, false);
      s1 = __builtin_amdgcn_wmma_f32_16x16x32_bf16(false, qf.v, false, kf1.v,
                                                   (short)0, s1, false, false);
    }

    // ---- causal mask + online softmax (rows live in 16-lane half-waves) ----
    float corr[8];
#pragma unroll
    for (int r = 0; r < 8; ++r) {
      int qrow = qr0 + mrow + r;
      float v0 = s0[r]; if (kt * 32 + lr > qrow)      v0 = -1e30f;
      float v1 = s1[r]; if (kt * 32 + 16 + lr > qrow) v1 = -1e30f;
      float t = fmaxf(v0, v1);
#pragma unroll
      for (int off = 1; off < 16; off <<= 1) t = fmaxf(t, __shfl_xor(t, off, 32));
      float mnew = fmaxf(row_max[r], t);
      float c = __expf(row_max[r] - mnew);
      float p0 = __expf(v0 - mnew);
      float p1 = __expf(v1 - mnew);
      float ps = p0 + p1;
#pragma unroll
      for (int off = 1; off < 16; off <<= 1) ps += __shfl_xor(ps, off, 32);
      row_sum[r] = row_sum[r] * c + ps;
      row_max[r] = mnew;
      corr[r] = c;
      Ps[wid][(mrow + r) * 40 + lr]      = f2bf(p0);
      Ps[wid][(mrow + r) * 40 + 16 + lr] = f2bf(p1);
    }
#pragma unroll
    for (int ft = 0; ft < 4; ++ft)
#pragma unroll
      for (int r = 0; r < 8; ++r) o[ft][r] *= corr[r];
    __syncthreads();  // Ps visible (and acts as LDS ordering fence)

    // ---- O[16 x 64] += P[16 x 32] * V[32 x 64] ----
    Frag pf;
    int pb = lr * 40 + kb;
    pf.q[0] = *(const uint4*)&Ps[wid][pb];
    pf.q[1] = *(const uint4*)&Ps[wid][pb + 16];
#pragma unroll
    for (int ft = 0; ft < 4; ++ft) {
      Frag vf;
      int vb = (ft * 16 + lr) * 40 + kb;
      vf.q[0] = *(const uint4*)&Vt[vb];
      vf.q[1] = *(const uint4*)&Vt[vb + 16];
      o[ft] = __builtin_amdgcn_wmma_f32_16x16x32_bf16(false, pf.v, false, vf.v,
                                                      (short)0, o[ft], false, false);
    }
  }

  // ---- normalize and write bf16 attention output [token][1024] ----
#pragma unroll
  for (int ft = 0; ft < 4; ++ft) {
#pragma unroll
    for (int r = 0; r < 8; ++r) {
      int qrow = qr0 + mrow + r;
      size_t tok = (size_t)b * SEQ + qrow;
      out[tok * DMODEL + h * 64 + ft * 16 + lr] = f2bf(o[ft][r] / row_sum[r]);
    }
  }
}

// ---------------------------------------------------------------------------
// Host launcher
// ---------------------------------------------------------------------------
extern "C" void kernel_launch(void* const* d_in, const int* in_sizes, int n_in,
                              void* d_out, int out_size, void* d_ws, size_t ws_size,
                              hipStream_t stream) {
  (void)in_sizes; (void)n_in; (void)out_size; (void)ws_size;
  const float* x      = (const float*)d_in[0];
  const float* w_qkv  = (const float*)d_in[1];
  const float* w_o    = (const float*)d_in[2];
  const float* w_gate = (const float*)d_in[3];
  const float* w_up   = (const float*)d_in[4];
  const float* w_down = (const float*)d_in[5];
  const float* g1     = (const float*)d_in[6];
  const float* g2     = (const float*)d_in[7];
  float* out = (float*)d_out;

  char* ws = (char*)d_ws;
  size_t off = 0;
  auto alloc = [&](size_t bytes) -> char* {
    char* p = ws + off;
    off += (bytes + 255) & ~(size_t)255;
    return p;
  };
  bfraw* wqkvT  = (bfraw*)alloc((size_t)NQKV   * DMODEL * 2);
  bfraw* woT    = (bfraw*)alloc((size_t)DMODEL * DMODEL * 2);
  bfraw* wgateT = (bfraw*)alloc((size_t)DFF    * DMODEL * 2);
  bfraw* wupT   = (bfraw*)alloc((size_t)DFF    * DMODEL * 2);
  bfraw* wdownT = (bfraw*)alloc((size_t)DMODEL * DFF    * 2);
  bfraw* xn1    = (bfraw*)alloc((size_t)TOKENS * DMODEL * 2);
  bfraw* qkv    = (bfraw*)alloc((size_t)TOKENS * NQKV   * 2);
  bfraw* attnb  = (bfraw*)alloc((size_t)TOKENS * DMODEL * 2);
  float* h1     = (float*)alloc((size_t)TOKENS * DMODEL * 4);
  bfraw* xn2    = (bfraw*)alloc((size_t)TOKENS * DMODEL * 2);
  float* gbuf   = (float*)alloc((size_t)TOKENS * DFF    * 4);
  bfraw* hmid   = (bfraw*)alloc((size_t)TOKENS * DFF    * 2);

  dim3 blk(256);

  // Weight cast+transpose: fp32 [K][N] -> bf16 [N][K]
  transpose_cast_kernel<<<dim3(NQKV / 32, DMODEL / 32), blk, 0, stream>>>(w_qkv,  wqkvT,  DMODEL, NQKV);
  transpose_cast_kernel<<<dim3(DMODEL / 32, DMODEL / 32), blk, 0, stream>>>(w_o,   woT,    DMODEL, DMODEL);
  transpose_cast_kernel<<<dim3(DFF / 32, DMODEL / 32), blk, 0, stream>>>(w_gate, wgateT, DMODEL, DFF);
  transpose_cast_kernel<<<dim3(DFF / 32, DMODEL / 32), blk, 0, stream>>>(w_up,   wupT,   DMODEL, DFF);
  transpose_cast_kernel<<<dim3(DMODEL / 32, DFF / 32), blk, 0, stream>>>(w_down, wdownT, DFF, DMODEL);

  // x1 = rmsnorm(x, g1) -> bf16
  rmsnorm_kernel<<<TOKENS, blk, 0, stream>>>(x, g1, xn1);
  // qkv = x1 @ w_qkv -> bf16
  gemm_bf16_kernel<MODE_BF16><<<dim3(NQKV / 128, TOKENS / 128), blk, 0, stream>>>(
      xn1, wqkvT, nullptr, nullptr, nullptr, qkv, TOKENS, NQKV, DMODEL);
  // attention -> bf16
  attn_kernel<<<dim3(SEQ / 128, 32), blk, 0, stream>>>(qkv, attnb);
  // h1 = x + attn @ w_o -> fp32
  gemm_bf16_kernel<MODE_RES_F32><<<dim3(DMODEL / 128, TOKENS / 128), blk, 0, stream>>>(
      attnb, woT, x, nullptr, h1, nullptr, TOKENS, DMODEL, DMODEL);
  // x2 = rmsnorm(h1, g2) -> bf16
  rmsnorm_kernel<<<TOKENS, blk, 0, stream>>>(h1, g2, xn2);
  // gbuf = x2 @ w_gate -> fp32
  gemm_bf16_kernel<MODE_F32><<<dim3(DFF / 128, TOKENS / 128), blk, 0, stream>>>(
      xn2, wgateT, nullptr, nullptr, gbuf, nullptr, TOKENS, DFF, DMODEL);
  // hmid = silu(gbuf) * (x2 @ w_up) -> bf16
  gemm_bf16_kernel<MODE_SILU_BF16><<<dim3(DFF / 128, TOKENS / 128), blk, 0, stream>>>(
      xn2, wupT, nullptr, gbuf, nullptr, hmid, TOKENS, DFF, DMODEL);
  // out = h1 + hmid @ w_down -> fp32
  gemm_bf16_kernel<MODE_RES_F32><<<dim3(DMODEL / 128, TOKENS / 128), blk, 0, stream>>>(
      hmid, wdownT, h1, nullptr, out, nullptr, TOKENS, DMODEL, DFF);
}
